// DirectedGraphConvolution_71597104824915
// MI455X (gfx1250) — matrix operand order repaired and checked by reading, not compiled, not run on hardware
//
#include <hip/hip_runtime.h>

// ---------------------------------------------------------------------------
// Types for CDNA5 WMMA
// ---------------------------------------------------------------------------
typedef __attribute__((ext_vector_type(16))) __bf16 v16bf;
typedef __attribute__((ext_vector_type(8)))  float  v8f;

union FragBF {
  v16bf v;
  uint4 q[2];   // 2 x 16B = 16 bf16
};

#define BLK_M 128
#define BLK_N 128
#define BLK_K 32
#define LDS_LDA 40   // 32 + 8 bf16 pad -> 80B row stride, 16B aligned, bank-spread
#define LDS_LDB 40

// Split two f32 into packed-bf16 hi words and packed-bf16 lo (residual) words.
__device__ __forceinline__ void packsplit2(float x, float y,
                                           unsigned int& hi, unsigned int& lo) {
  union { __bf16 b[2]; unsigned int u; } H, L;
  H.b[0] = (__bf16)x;
  H.b[1] = (__bf16)y;
  L.b[0] = (__bf16)(x - (float)H.b[0]);
  L.b[1] = (__bf16)(y - (float)H.b[1]);
  hi = H.u;
  lo = L.u;
}

// ---------------------------------------------------------------------------
// Generic strided-batch GEMM:  C[b] (+relu) = A[b] @ (diag(bscale[b]) B[b])
//   A: f32, strides (sAm, sAk)  -> transpose == swap strides
//   B: f32, row-major K x N, optional per-k-row scale vector
//   C: f32, row stride ldc (write directly into 768-wide concat output)
// bf16x3 split accumulation on v_wmma_f32_16x16x32_bf16 (hh + lh + hl).
// Requires M%128==0, N%128==0, K%32==0 (true for all shapes here).
// Block: 256 threads = 8 waves (4 along M, 2 along N); wave tile 32x64.
// Register-prefetch pipeline with 32-bit incremented element offsets so the
// backend can use SADDR(base)+VADDR(offset) addressing (no per-chunk mul64).
// ---------------------------------------------------------------------------
__global__ __launch_bounds__(256) void gemm_bf16x3(
    const float* __restrict__ A, long long sAb, int sAm, int sAk,
    const float* __restrict__ B, long long sBb, int ldb,
    const float* __restrict__ bscale, int sSb,
    float* __restrict__ C, long long sCb, int ldc,
    int K, int do_relu)
{
  __shared__ __bf16 sAhi[BLK_M * LDS_LDA];
  __shared__ __bf16 sAlo[BLK_M * LDS_LDA];
  __shared__ __bf16 sBhi[BLK_N * LDS_LDB];
  __shared__ __bf16 sBlo[BLK_N * LDS_LDB];

  const int tid   = threadIdx.x;
  const int lane  = tid & 31;
  const int wave  = tid >> 5;
  const int wm0   = (wave & 3) * 32;   // 4 waves along M
  const int wn0   = (wave >> 2) * 64;  // 2 waves along N
  const int lhalf = lane >> 4;         // 0 or 1 (half-wave)
  const int lmod  = lane & 15;

  const int m0 = blockIdx.x * BLK_M;
  const int n0 = blockIdx.y * BLK_N;
  const int b  = blockIdx.z;

  const float* Ab = A + (size_t)b * (size_t)sAb;
  const float* Bb = B + (size_t)b * (size_t)sBb;
  float*       Cb = C + (size_t)b * (size_t)sCb;
  const float* Sb = bscale ? (bscale + (size_t)b * (size_t)sSb) : nullptr;

  const bool a_rowmajor = (sAk == 1);

  // thread->element mappings (fixed per thread)
  const int ar = tid >> 3;             // row-major A: base row 0..31
  const int ac = (tid & 7) << 2;       // row-major A: k offset 0..28
  const int mq = lane << 2;            // transA: m offset 0..124
  const int nq = lane << 2;            // B: n offset 0..124
  const int kp = (tid >> 5) << 1;      // transA/B: k pair base 0..14

  // 32-bit element offsets, advanced by a fixed step per K-chunk
  unsigned int offA[4], offB[4], offS;
  unsigned int stepA, stepB;
  if (a_rowmajor) {
    stepA = BLK_K;
    #pragma unroll
    for (int r = 0; r < 4; ++r)
      offA[r] = (unsigned int)((m0 + ar + r * 32) * sAm + ac);
  } else {
    stepA = (unsigned int)(BLK_K * sAk);
    #pragma unroll
    for (int r = 0; r < 2; ++r) {
      offA[2 * r]     = (unsigned int)((kp + r * 16) * sAk + m0 + mq);
      offA[2 * r + 1] = offA[2 * r] + (unsigned int)sAk;
    }
  }
  stepB = (unsigned int)(BLK_K * ldb);
  #pragma unroll
  for (int r = 0; r < 2; ++r) {
    offB[2 * r]     = (unsigned int)((kp + r * 16) * ldb + n0 + nq);
    offB[2 * r + 1] = offB[2 * r] + (unsigned int)ldb;
  }
  offS = (unsigned int)kp;

  float4 ra[4];
  float4 rb[4];
  float  rs0[2], rs1[2];

  auto prefetch = [&]() {
    #pragma unroll
    for (int r = 0; r < 4; ++r) {
      ra[r] = *(const float4*)(Ab + offA[r]);
      offA[r] += stepA;
    }
    #pragma unroll
    for (int r = 0; r < 4; ++r) {
      rb[r] = *(const float4*)(Bb + offB[r]);
      offB[r] += stepB;
    }
    if (Sb) {
      rs0[0] = Sb[offS];      rs1[0] = Sb[offS + 1];
      rs0[1] = Sb[offS + 16]; rs1[1] = Sb[offS + 17];
    } else {
      rs0[0] = rs0[1] = rs1[0] = rs1[1] = 1.f;
    }
    offS += BLK_K;
  };

  auto stage = [&]() {
    if (a_rowmajor) {
      #pragma unroll
      for (int r = 0; r < 4; ++r) {
        int base = (ar + r * 32) * LDS_LDA + ac;
        uint2 h, l;
        packsplit2(ra[r].x, ra[r].y, h.x, l.x);
        packsplit2(ra[r].z, ra[r].w, h.y, l.y);
        *(uint2*)&sAhi[base] = h;
        *(uint2*)&sAlo[base] = l;
      }
    } else {
      // pack pairs along k: {A[m][kl], A[m][kl+1]} -> one b32 store per plane
      #pragma unroll
      for (int r = 0; r < 2; ++r) {
        int kl = kp + r * 16;
        float4 v0 = ra[2 * r], v1 = ra[2 * r + 1];
        unsigned int h, l;
        packsplit2(v0.x, v1.x, h, l);
        *(unsigned int*)&sAhi[(mq + 0) * LDS_LDA + kl] = h;
        *(unsigned int*)&sAlo[(mq + 0) * LDS_LDA + kl] = l;
        packsplit2(v0.y, v1.y, h, l);
        *(unsigned int*)&sAhi[(mq + 1) * LDS_LDA + kl] = h;
        *(unsigned int*)&sAlo[(mq + 1) * LDS_LDA + kl] = l;
        packsplit2(v0.z, v1.z, h, l);
        *(unsigned int*)&sAhi[(mq + 2) * LDS_LDA + kl] = h;
        *(unsigned int*)&sAlo[(mq + 2) * LDS_LDA + kl] = l;
        packsplit2(v0.w, v1.w, h, l);
        *(unsigned int*)&sAhi[(mq + 3) * LDS_LDA + kl] = h;
        *(unsigned int*)&sAlo[(mq + 3) * LDS_LDA + kl] = l;
      }
    }
    #pragma unroll
    for (int r = 0; r < 2; ++r) {
      int kl = kp + r * 16;
      float4 v0 = rb[2 * r], v1 = rb[2 * r + 1];
      float s0 = rs0[r], s1 = rs1[r];
      v0.x *= s0; v0.y *= s0; v0.z *= s0; v0.w *= s0;
      v1.x *= s1; v1.y *= s1; v1.z *= s1; v1.w *= s1;
      unsigned int h, l;
      packsplit2(v0.x, v1.x, h, l);
      *(unsigned int*)&sBhi[(nq + 0) * LDS_LDB + kl] = h;
      *(unsigned int*)&sBlo[(nq + 0) * LDS_LDB + kl] = l;
      packsplit2(v0.y, v1.y, h, l);
      *(unsigned int*)&sBhi[(nq + 1) * LDS_LDB + kl] = h;
      *(unsigned int*)&sBlo[(nq + 1) * LDS_LDB + kl] = l;
      packsplit2(v0.z, v1.z, h, l);
      *(unsigned int*)&sBhi[(nq + 2) * LDS_LDB + kl] = h;
      *(unsigned int*)&sBlo[(nq + 2) * LDS_LDB + kl] = l;
      packsplit2(v0.w, v1.w, h, l);
      *(unsigned int*)&sBhi[(nq + 3) * LDS_LDB + kl] = h;
      *(unsigned int*)&sBlo[(nq + 3) * LDS_LDB + kl] = l;
    }
  };

  const v8f vzero = {0.f, 0.f, 0.f, 0.f, 0.f, 0.f, 0.f, 0.f};
  v8f acc[2][4];
  #pragma unroll
  for (int mi = 0; mi < 2; ++mi)
    #pragma unroll
    for (int ni = 0; ni < 4; ++ni) acc[mi][ni] = vzero;

  prefetch();

  for (int kb = 0; kb < K; kb += BLK_K) {
    if (kb) __syncthreads();   // previous chunk's LDS reads done
    stage();
    __syncthreads();

    if (kb + BLK_K < K) prefetch();  // overlap global loads with WMMA below

    // ---- fragments per ISA 16-bit WMMA layouts ----
    FragBF ah[2], al[2], bh[4], bl[4];
    #pragma unroll
    for (int mi = 0; mi < 2; ++mi) {
      // A 16x32: lane<16 -> M=lane, K={0..7,16..23}; lane>=16 -> K={8..15,24..31}
      int row = wm0 + mi * 16 + lmod;
      int off = row * LDS_LDA + lhalf * 8;
      ah[mi].q[0] = *(const uint4*)&sAhi[off];
      ah[mi].q[1] = *(const uint4*)&sAhi[off + 16];
      al[mi].q[0] = *(const uint4*)&sAlo[off];
      al[mi].q[1] = *(const uint4*)&sAlo[off + 16];
    }
    #pragma unroll
    for (int ni = 0; ni < 4; ++ni) {
      // B 32x16: lane<16 -> N=lane, K={0..7,16..23}; lane>=16 -> K={8..15,24..31}
      int col = wn0 + ni * 16 + lmod;
      int off = col * LDS_LDB + lhalf * 8;
      bh[ni].q[0] = *(const uint4*)&sBhi[off];
      bh[ni].q[1] = *(const uint4*)&sBhi[off + 16];
      bl[ni].q[0] = *(const uint4*)&sBlo[off];
      bl[ni].q[1] = *(const uint4*)&sBlo[off + 16];
    }

    // ---- bf16x3 accumulation: hh + lh + hl (drop ll, ~2^-16 rel err) ----
    #pragma unroll
    for (int mi = 0; mi < 2; ++mi)
      #pragma unroll
      for (int ni = 0; ni < 4; ++ni) {
        acc[mi][ni] = __builtin_amdgcn_wmma_f32_16x16x32_bf16(
            false, ah[mi].v, false, bh[ni].v, (short)0, acc[mi][ni], false, false);
        acc[mi][ni] = __builtin_amdgcn_wmma_f32_16x16x32_bf16(
            false, al[mi].v, false, bh[ni].v, (short)0, acc[mi][ni], false, false);
        acc[mi][ni] = __builtin_amdgcn_wmma_f32_16x16x32_bf16(
            false, ah[mi].v, false, bl[ni].v, (short)0, acc[mi][ni], false, false);
      }
  }

  // ---- epilogue: C/D layout: VGPR v, lane<16 -> M=v ; lane>=16 -> M=v+8 ----
  #pragma unroll
  for (int mi = 0; mi < 2; ++mi) {
    int rbase = m0 + wm0 + mi * 16 + lhalf * 8;
    #pragma unroll
    for (int ni = 0; ni < 4; ++ni) {
      int col = n0 + wn0 + ni * 16 + lmod;
      #pragma unroll
      for (int v = 0; v < 8; ++v) {
        float x = acc[mi][ni][v];
        if (do_relu) x = fmaxf(x, 0.0f);
        Cb[(size_t)(rbase + v) * (size_t)ldc + (size_t)col] = x;
      }
    }
  }
}

// ---------------------------------------------------------------------------
// Helper kernels
// ---------------------------------------------------------------------------
__global__ void zero_f32(float* p, int n) {
  int i = blockIdx.x * blockDim.x + threadIdx.x;
  if (i < n) p[i] = 0.f;
}

// one block per (b,row): rsinv[b*N+i] = 1 / sum_j G[b][i][j]
__global__ void rowsum_inv_kernel(const float* __restrict__ G, float* __restrict__ rsinv, int N) {
  __shared__ float red[256];
  const float* row = G + (size_t)blockIdx.x * (size_t)N;
  float s = 0.f;
  for (int j = threadIdx.x * 4; j < N; j += blockDim.x * 4) {
    float4 v = *(const float4*)(row + j);
    s += v.x + v.y + v.z + v.w;
  }
  red[threadIdx.x] = s;
  __syncthreads();
  for (int o = 128; o > 0; o >>= 1) {
    if ((int)threadIdx.x < o) red[threadIdx.x] += red[threadIdx.x + o];
    __syncthreads();
  }
  if (threadIdx.x == 0) rsinv[blockIdx.x] = 1.0f / red[0];
}

// partial column sums with f32 global atomics (csum must be zeroed first)
__global__ void colsum_partial_kernel(const float* __restrict__ G, float* __restrict__ csum,
                                      int N, int rows) {
  int b = blockIdx.z;
  int j = blockIdx.x * blockDim.x + threadIdx.x;
  int r0 = blockIdx.y * rows;
  const float* Gb = G + (size_t)b * (size_t)N * (size_t)N;
  float s = 0.f;
  for (int k = 0; k < rows; ++k) s += Gb[(size_t)(r0 + k) * (size_t)N + (size_t)j];
  atomicAdd(&csum[b * N + j], s);
}

__global__ void invert_kernel(const float* __restrict__ in, float* __restrict__ out, int n) {
  int i = blockIdx.x * blockDim.x + threadIdx.x;
  if (i < n) out[i] = 1.0f / in[i];
}

// out[:, 0:256] = relu(0.5*(P+Q)), output row stride 768
__global__ void out1_kernel(const float* __restrict__ P, const float* __restrict__ Q,
                            float* __restrict__ out, int nvec) {
  int i = blockIdx.x * blockDim.x + threadIdx.x;
  if (i >= nvec) return;
  float4 p = ((const float4*)P)[i];
  float4 q = ((const float4*)Q)[i];
  float4 o;
  o.x = fmaxf(0.f, 0.5f * (p.x + q.x));
  o.y = fmaxf(0.f, 0.5f * (p.y + q.y));
  o.z = fmaxf(0.f, 0.5f * (p.z + q.z));
  o.w = fmaxf(0.f, 0.5f * (p.w + q.w));
  int e0 = i << 2;
  int row = e0 >> 8;      // DOUT = 256
  int d = e0 & 255;
  *(float4*)(out + (size_t)row * 768 + (size_t)d) = o;
}

// ---------------------------------------------------------------------------
// Launch:
//   rs,cs sums ; Ne = H@W ; P = G@Ne ; Q = G^T@Ne ;
//   out[:,0:256]   = relu(0.5(P+Q))
//   out[:,256:512] = relu(G^T @ diag(1/rs) P)
//   out[:,512:768] = relu(G   @ diag(1/cs) Q)
// Workspace: Ne,P,Q (3 x 16MB) + rsinv/csum/csinv (~192KB) ~= 48.5 MB
// ---------------------------------------------------------------------------
extern "C" void kernel_launch(void* const* d_in, const int* in_sizes, int n_in,
                              void* d_out, int out_size, void* d_ws, size_t ws_size,
                              hipStream_t stream) {
  (void)in_sizes; (void)n_in; (void)out_size; (void)ws_size;
  const int B = 8, N = 2048, DIN = 256, DOUT = 256;
  const float* H = (const float*)d_in[0];
  const float* G = (const float*)d_in[1];
  const float* W = (const float*)d_in[2];
  float* out = (float*)d_out;
  float* ws  = (float*)d_ws;

  const size_t NE = (size_t)B * N * DOUT;   // 4,194,304 floats
  float* Ne    = ws;
  float* P     = ws + NE;
  float* Q     = ws + 2 * NE;
  float* rsinv = ws + 3 * NE;
  float* csum  = rsinv + (size_t)B * N;
  float* csinv = csum + (size_t)B * N;

  const long long NN = (long long)N * N;
  const long long ND = (long long)N * DOUT;
  const long long NO = (long long)N * 3 * DOUT;

  // --- row/col sum reciprocals ---
  zero_f32<<<(B * N + 255) / 256, 256, 0, stream>>>(csum, B * N);
  rowsum_inv_kernel<<<B * N, 256, 0, stream>>>(G, rsinv, N);
  colsum_partial_kernel<<<dim3(N / 256, 8, B), 256, 0, stream>>>(G, csum, N, N / 8);
  invert_kernel<<<(B * N + 255) / 256, 256, 0, stream>>>(csum, csinv, B * N);

  // --- Ne = H @ W  (M=16384, N=256, K=256, single "batch") ---
  gemm_bf16x3<<<dim3((B * N) / BLK_M, DOUT / BLK_N, 1), 256, 0, stream>>>(
      H, 0, DIN, 1, W, 0, DOUT, nullptr, 0, Ne, 0, DOUT, DIN, 0);

  // --- P = G @ Ne ---
  gemm_bf16x3<<<dim3(N / BLK_M, DOUT / BLK_N, B), 256, 0, stream>>>(
      G, NN, N, 1, Ne, ND, DOUT, nullptr, 0, P, ND, DOUT, N, 0);

  // --- Q = G^T @ Ne ---
  gemm_bf16x3<<<dim3(N / BLK_M, DOUT / BLK_N, B), 256, 0, stream>>>(
      G, NN, 1, N, Ne, ND, DOUT, nullptr, 0, Q, ND, DOUT, N, 0);

  // --- out[:, 0:256] = relu(0.5*(P+Q)) ---
  out1_kernel<<<(int)(NE / 4 / 256), 256, 0, stream>>>(P, Q, out, (int)(NE / 4));

  // --- out[:, 256:512] = relu(G^T @ diag(1/rs) P) ---
  gemm_bf16x3<<<dim3(N / BLK_M, DOUT / BLK_N, B), 256, 0, stream>>>(
      G, NN, 1, N, P, ND, DOUT, rsinv, N, out + DOUT, NO, 3 * DOUT, N, 1);

  // --- out[:, 512:768] = relu(G @ diag(1/cs) Q) ---
  gemm_bf16x3<<<dim3(N / BLK_M, DOUT / BLK_N, B), 256, 0, stream>>>(
      G, NN, N, 1, Q, ND, DOUT, csinv, N, out + 2 * DOUT, NO, 3 * DOUT, N, 1);
}
